// MaskedMultiHeadAttentionParallel_19756849561619
// MI455X (gfx1250) — matrix operand order repaired and checked by reading, not compiled, not run on hardware
//
#include <hip/hip_runtime.h>
#include <hip/hip_bf16.h>

// Problem sizes (fixed by the reference)
#define HH 16
#define DD 1024
#define KD 64
#define NN 4096
#define MM 4096
#define MW (MM / 32)   // packed mask words per row

typedef __attribute__((ext_vector_type(16))) __bf16 v16bf;
typedef __attribute__((ext_vector_type(8)))  float  v8f;

union V16U { v16bf v; uint4 q[2]; };

static __device__ __forceinline__ unsigned short f2bf(float f) {
  unsigned int u = __float_as_uint(f);
  u = (u + 0x7FFFu + ((u >> 16) & 1u)) >> 16;   // round-to-nearest-even
  return (unsigned short)u;
}

// 16 contiguous bf16 (32B) -> one v16bf (B-operand layout: per-lane contiguous K run)
static __device__ __forceinline__ v16bf load16_contig(const unsigned short* p) {
  V16U u;
  u.q[0] = *reinterpret_cast<const uint4*>(p);
  u.q[1] = *reinterpret_cast<const uint4*>(p + 8);
  return u.v;
}

// A-operand layout: 16 contiguous bf16 at p (per-lane K runs 0..15 / 16..31)
static __device__ __forceinline__ v16bf load_a_halves(const unsigned short* p) {
  V16U u;
  u.q[0] = *reinterpret_cast<const uint4*>(p);
  u.q[1] = *reinterpret_cast<const uint4*>(p + 8);
  return u.v;
}

// ---------------- Stage 1: conversions / transposes / mask packing ----------------

__global__ void k_f32_to_bf16(const float* __restrict__ src,
                              unsigned short* __restrict__ dst, int n) {
  int i = blockIdx.x * blockDim.x + threadIdx.x;
  int stride = gridDim.x * blockDim.x;
  for (; i < n; i += stride) dst[i] = f2bf(src[i]);
}

// mask int32 [N][M] -> bit-packed [N][M/32] (bit j of word w = mask[r][32w+j])
// 64 MB of int32 -> 2 MB of bits: stays L2-resident across all 16 head passes.
__global__ void k_pack_mask(const int* __restrict__ mask,
                            unsigned* __restrict__ mp) {
  int i = blockIdx.x * blockDim.x + threadIdx.x;
  const int total = NN * MW;
  int stride = gridDim.x * blockDim.x;
  for (; i < total; i += stride) {
    int r = i / MW;
    int w = i % MW;
    const int* src = mask + (size_t)r * MM + w * 32;
    unsigned bits = 0;
#pragma unroll
    for (int j = 0; j < 32; ++j) bits |= (src[j] != 0 ? 1u : 0u) << j;
    mp[i] = bits;
  }
}

// W: [H][D][64] f32  ->  WT: [H][64][D] bf16  (so B loads are contiguous in K=D)
__global__ void k_transpose_w(const float* __restrict__ W,
                              unsigned short* __restrict__ WT) {
  int i = blockIdx.x * blockDim.x + threadIdx.x;
  const int total = HH * DD * KD;
  int stride = gridDim.x * blockDim.x;
  for (; i < total; i += stride) {
    int c = i % KD;
    int d = (i / KD) % DD;
    int h = i / (KD * DD);
    WT[((size_t)h * KD + c) * DD + d] = f2bf(W[i]);
  }
}

// Wo: [H][64][D] f32 -> WoT: [D][H*64] bf16 (B operand of the fused output GEMM)
__global__ void k_transpose_wo(const float* __restrict__ Wo,
                               unsigned short* __restrict__ WoT) {
  int i = blockIdx.x * blockDim.x + threadIdx.x;
  const int total = HH * KD * DD;
  int stride = gridDim.x * blockDim.x;
  for (; i < total; i += stride) {
    int d = i % DD;
    int v = (i / DD) % KD;
    int h = i / (DD * KD);
    WoT[(size_t)d * (HH * KD) + h * KD + v] = f2bf(Wo[i]);
  }
}

// ---------------- Stage 2: Q/K/V projections (WMMA GEMM) ----------------
// A: [rows][D] bf16, WT: [H][64][D] bf16.
// transposeOut==0 -> out[H][rows][64]; transposeOut==1 -> out[H][64][rows]
__global__ void k_proj(const unsigned short* __restrict__ A,
                       const unsigned short* __restrict__ WT,
                       unsigned short* __restrict__ out,
                       int rows, int transposeOut) {
  const int lane = threadIdx.x & 31;
  const int wave = threadIdx.x >> 5;       // 4 waves -> 4 column tiles of 16 (64 cols)
  const int l16  = lane & 15;
  const int half = lane >> 4;
  const int tilesPerHead = rows / 16;
  const int h     = blockIdx.x / tilesPerHead;
  const int rtile = blockIdx.x % tilesPerHead;
  const int row   = rtile * 16 + l16;
  const int col   = wave * 16 + l16;

  const unsigned short* arow = A + (size_t)row * DD;
  const unsigned short* bcol = WT + ((size_t)h * KD + col) * DD;

  v8f acc = {};
  for (int k0 = 0; k0 < DD; k0 += 32) {
    __builtin_prefetch(arow + k0 + 256, 0, 1);
    __builtin_prefetch(bcol + k0 + 256, 0, 1);
    v16bf a = load_a_halves(arow + k0 + half * 16);
    v16bf b = load16_contig(bcol + k0 + half * 16);
    acc = __builtin_amdgcn_wmma_f32_16x16x32_bf16(false, a, false, b,
                                                  (short)0, acc, false, false);
  }

#pragma unroll
  for (int v = 0; v < 8; ++v) {
    int r = rtile * 16 + v + 8 * half;     // C layout: VGPR v holds rows v / v+8
    unsigned short val = f2bf(acc[v]);
    if (!transposeOut)
      out[((size_t)h * rows + r) * KD + col] = val;
    else
      out[((size_t)h * KD + col) * rows + r] = val;
  }
}

// ---------------- Stage 3: flash attention (one wave per 16 queries) ----------------
__global__ void k_attn(const unsigned short* __restrict__ Q,     // [H][N][64]
                       const unsigned short* __restrict__ K,     // [H][M][64]
                       const unsigned short* __restrict__ VT,    // [H][64][M]
                       const unsigned* __restrict__ maskP,       // [N][M/32] bits
                       unsigned short* __restrict__ Ocat) {      // [N][H*64]
  __shared__ __align__(16) unsigned short Plds[16 * 32];

  const int lane  = threadIdx.x & 31;
  const int l16   = lane & 15;
  const int half  = lane >> 4;
  const int tiles = NN / 16;
  const int h     = blockIdx.x / tiles;
  const int ntile = blockIdx.x % tiles;
  const int qrow  = ntile * 16 + l16;

  // Q tile in A layout, both 32-wide K chunks, loaded once
  const unsigned short* qp = Q + ((size_t)h * NN + qrow) * KD;
  v16bf qa0 = load_a_halves(qp + 0  + half * 16);
  v16bf qa1 = load_a_halves(qp + 32 + half * 16);

  float mi[8], li[8];
  v8f acc[4] = {};
#pragma unroll
  for (int v = 0; v < 8; ++v) { mi[v] = -1e30f; li[v] = 0.0f; }

  const unsigned short* kbase = K  + (size_t)h * MM * KD;
  const unsigned short* vbase = VT + (size_t)h * KD * MM;

  for (int mb = 0; mb < MM; mb += 32) {
    const int mword = mb >> 5;

    // ---- logits = Q * K^T : two 16x16 C tiles (4 WMMAs) ----
    const unsigned short* k0p = kbase + (size_t)(mb + l16) * KD;
    const unsigned short* k1p = kbase + (size_t)(mb + 16 + l16) * KD;
    v16bf b00 = load16_contig(k0p + half * 16);       // m-tile 0, features 0..31
    v16bf b01 = load16_contig(k0p + 32 + half * 16);  // m-tile 0, features 32..63
    v16bf b10 = load16_contig(k1p + half * 16);
    v16bf b11 = load16_contig(k1p + 32 + half * 16);

    v8f c0 = {};
    c0 = __builtin_amdgcn_wmma_f32_16x16x32_bf16(false, qa0, false, b00, (short)0, c0, false, false);
    c0 = __builtin_amdgcn_wmma_f32_16x16x32_bf16(false, qa1, false, b01, (short)0, c0, false, false);
    v8f c1 = {};
    c1 = __builtin_amdgcn_wmma_f32_16x16x32_bf16(false, qa0, false, b10, (short)0, c1, false, false);
    c1 = __builtin_amdgcn_wmma_f32_16x16x32_bf16(false, qa1, false, b11, (short)0, c1, false, false);

    // ---- mask (one packed word per row covers the whole 32-key block) ----
    // ---- + online softmax (row stats via intra-half shfl_xor) ----
    float p0[8], p1[8], scale[8];
#pragma unroll
    for (int v = 0; v < 8; ++v) {
      int r = ntile * 16 + v + 8 * half;
      unsigned w = maskP[(size_t)r * MW + mword];
      float x0 = ((w >> l16) & 1u)        ? c0[v] : -1e30f;
      float x1 = ((w >> (16 + l16)) & 1u) ? c1[v] : -1e30f;
      float rm = fmaxf(x0, x1);
      rm = fmaxf(rm, __shfl_xor(rm, 1));
      rm = fmaxf(rm, __shfl_xor(rm, 2));
      rm = fmaxf(rm, __shfl_xor(rm, 4));
      rm = fmaxf(rm, __shfl_xor(rm, 8));   // masks <16 stay inside the 16-lane half
      float mn = fmaxf(mi[v], rm);
      float sc = __expf(mi[v] - mn);
      float e0 = __expf(x0 - mn);
      float e1 = __expf(x1 - mn);
      float s  = e0 + e1;
      s += __shfl_xor(s, 1);
      s += __shfl_xor(s, 2);
      s += __shfl_xor(s, 4);
      s += __shfl_xor(s, 8);
      li[v] = li[v] * sc + s;
      mi[v] = mn;
      p0[v] = e0; p1[v] = e1; scale[v] = sc;
    }
#pragma unroll
    for (int j = 0; j < 4; ++j)
#pragma unroll
      for (int v = 0; v < 8; ++v) acc[j][v] *= scale[v];

    // ---- C layout -> A layout for P via LDS (16x32 bf16 tile, row-major) ----
#pragma unroll
    for (int v = 0; v < 8; ++v) {
      Plds[(v + 8 * half) * 32 + l16]      = f2bf(p0[v]);
      Plds[(v + 8 * half) * 32 + 16 + l16] = f2bf(p1[v]);
    }
    __syncthreads();
    V16U pu;
    pu.q[0] = *reinterpret_cast<const uint4*>(&Plds[l16 * 32 + half * 16]);
    pu.q[1] = *reinterpret_cast<const uint4*>(&Plds[l16 * 32 + half * 16 + 8]);
    v16bf pa = pu.v;
    __syncthreads();

    // ---- O += P * V : 4 WMMAs over the four 16-wide v chunks ----
#pragma unroll
    for (int j = 0; j < 4; ++j) {
      const unsigned short* vp =
          vbase + (size_t)(j * 16 + l16) * MM + mb + half * 16;
      v16bf bv = load16_contig(vp);
      acc[j] = __builtin_amdgcn_wmma_f32_16x16x32_bf16(false, pa, false, bv,
                                                       (short)0, acc[j], false, false);
    }
  }

  // epilogue: normalize and emit head-concatenated O (bf16, A-operand friendly)
#pragma unroll
  for (int v = 0; v < 8; ++v) {
    float inv = 1.0f / fmaxf(li[v], 1e-30f);
    int r = ntile * 16 + v + 8 * half;
#pragma unroll
    for (int j = 0; j < 4; ++j)
      Ocat[(size_t)r * (HH * KD) + h * KD + j * 16 + l16] = f2bf(acc[j][v] * inv);
  }
}

// ---------------- Stage 4: fused output projection Y = Ocat @ Wo_cat ----------------
__global__ void k_out_gemm(const unsigned short* __restrict__ Ocat, // [N][1024]
                           const unsigned short* __restrict__ WoT,  // [1024 d][1024 hv]
                           float* __restrict__ Y) {                 // [N][1024] f32
  const int lane   = threadIdx.x & 31;
  const int wave   = threadIdx.x >> 5;    // 8 waves -> 128 output cols per block
  const int l16    = lane & 15;
  const int half   = lane >> 4;
  const int rtile  = blockIdx.x >> 3;
  const int cgroup = blockIdx.x & 7;
  const int col    = cgroup * 128 + wave * 16 + l16;
  const int row    = rtile * 16 + l16;

  const unsigned short* arow = Ocat + (size_t)row * DD;
  const unsigned short* bcol = WoT + (size_t)col * DD;

  v8f acc = {};
  for (int k0 = 0; k0 < DD; k0 += 32) {
    __builtin_prefetch(arow + k0 + 256, 0, 1);
    __builtin_prefetch(bcol + k0 + 256, 0, 1);
    v16bf a = load_a_halves(arow + k0 + half * 16);
    v16bf b = load16_contig(bcol + k0 + half * 16);
    acc = __builtin_amdgcn_wmma_f32_16x16x32_bf16(false, a, false, b,
                                                  (short)0, acc, false, false);
  }
#pragma unroll
  for (int v = 0; v < 8; ++v) {
    int r = rtile * 16 + v + 8 * half;
    Y[(size_t)r * DD + col] = acc[v];
  }
}

// ---------------- host launcher ----------------
extern "C" void kernel_launch(void* const* d_in, const int* in_sizes, int n_in,
                              void* d_out, int out_size, void* d_ws, size_t ws_size,
                              hipStream_t stream) {
  const float* X    = (const float*)d_in[0];
  const float* M    = (const float*)d_in[1];
  const int*   mask = (const int*)d_in[2];
  const float* Wq   = (const float*)d_in[3];
  const float* Wk   = (const float*)d_in[4];
  const float* Wv   = (const float*)d_in[5];
  const float* Wo   = (const float*)d_in[6];
  float* Y = (float*)d_out;

  // bump allocator over workspace (~58 MB total)
  char* ws = (char*)d_ws;
  auto alloc = [&](size_t bytes) -> void* {
    void* p = ws;
    ws += (bytes + 255) & ~(size_t)255;
    return p;
  };
  unsigned short* Xb    = (unsigned short*)alloc((size_t)NN * DD * 2);
  unsigned short* Mb    = (unsigned short*)alloc((size_t)MM * DD * 2);
  unsigned short* WqT   = (unsigned short*)alloc((size_t)HH * KD * DD * 2);
  unsigned short* WkT   = (unsigned short*)alloc((size_t)HH * KD * DD * 2);
  unsigned short* WvT   = (unsigned short*)alloc((size_t)HH * KD * DD * 2);
  unsigned short* WoT   = (unsigned short*)alloc((size_t)DD * HH * KD * 2);
  unsigned short* Qb    = (unsigned short*)alloc((size_t)HH * NN * KD * 2);
  unsigned short* Kb    = (unsigned short*)alloc((size_t)HH * MM * KD * 2);
  unsigned short* VTb   = (unsigned short*)alloc((size_t)HH * KD * MM * 2);
  unsigned short* Ocat  = (unsigned short*)alloc((size_t)NN * HH * KD * 2);
  unsigned*       maskP = (unsigned*)alloc((size_t)NN * MW * 4);

  k_f32_to_bf16<<<1024, 256, 0, stream>>>(X, Xb, NN * DD);
  k_f32_to_bf16<<<1024, 256, 0, stream>>>(M, Mb, MM * DD);
  k_pack_mask<<<1024, 256, 0, stream>>>(mask, maskP);
  k_transpose_w<<<1024, 256, 0, stream>>>(Wq, WqT);
  k_transpose_w<<<1024, 256, 0, stream>>>(Wk, WkT);
  k_transpose_w<<<1024, 256, 0, stream>>>(Wv, WvT);
  k_transpose_wo<<<1024, 256, 0, stream>>>(Wo, WoT);

  k_proj<<<HH * (NN / 16), 128, 0, stream>>>(Xb, WqT, Qb, NN, 0);
  k_proj<<<HH * (MM / 16), 128, 0, stream>>>(Mb, WkT, Kb, MM, 0);
  k_proj<<<HH * (MM / 16), 128, 0, stream>>>(Mb, WvT, VTb, MM, 1);

  k_attn<<<HH * (NN / 16), 32, 0, stream>>>(Qb, Kb, VTb, maskP, Ocat);

  k_out_gemm<<<(NN / 16) * 8, 256, 0, stream>>>(Ocat, WoT, Y);
}
